// CrossAttention_33663953666886
// MI455X (gfx1250) — matrix operand order repaired and checked by reading, not compiled, not run on hardware
//
#include <hip/hip_runtime.h>
#include <hip/hip_bf16.h>
#include <math.h>

// Problem constants (H is fixed by the model; B and N are taken from in_sizes).
#define HDIM   256
#define HDIM4  1024
#define INV_SCALE 0.17677669529663687f   // 1/sqrt(256/8)

typedef __attribute__((ext_vector_type(16))) __bf16 v16bf;
typedef __attribute__((ext_vector_type(8)))  float  v8f;
typedef __attribute__((ext_vector_type(4)))  unsigned int u32x4;
typedef __attribute__((ext_vector_type(8)))  int i32x8;
typedef __attribute__((ext_vector_type(4)))  int i32x4;

__device__ __forceinline__ float wave_reduce_sum(float v) {
#pragma unroll
  for (int o = 16; o > 0; o >>= 1) v += __shfl_xor(v, o, 32);
  return v;
}

// ---------------------------------------------------------------------------
// TDM: DMA a [32 rows x 64 cols] f32 tile of a row-major [nrows x rowlen]
// tensor into LDS (row-major, tightly packed 32x64). Descriptor built per the
// CDNA5 D# spec (group0: count/lds_addr/global_addr/type=2; group1:
// data_size=4B, tensor dims for OOB, tile 64x32, dim0 stride). 2-D tensor ->
// groups 2/3 are zero. Tracked by TENSORcnt.
// ---------------------------------------------------------------------------
__device__ __forceinline__ void tdm_load_b_tile(unsigned lds_addr,
                                                const float* gptr,
                                                int rowlen, int nrows) {
  const unsigned long long ga = (unsigned long long)(uintptr_t)gptr;
  u32x4 g0;
  g0[0] = 1u;                                           // count=1, user mode
  g0[1] = lds_addr;                                     // LDS byte address
  g0[2] = (unsigned)(ga & 0xFFFFFFFFu);                 // global_addr[31:0]
  g0[3] = (unsigned)((ga >> 32) & 0x1FFFFFFu) | (2u << 30); // addr[56:32] | type=2
  i32x8 g1;
  g1[0] = (int)(2u << 16);                              // data_size = 2 (4 bytes)
  g1[1] = (int)(((unsigned)rowlen & 0xFFFFu) << 16);    // tensor_dim0[15:0]
  g1[2] = (int)((((unsigned)rowlen >> 16) & 0xFFFFu) |  // tensor_dim0[31:16]
                (((unsigned)nrows & 0xFFFFu) << 16));   // tensor_dim1[15:0]
  g1[3] = (int)((((unsigned)nrows >> 16) & 0xFFFFu) |   // tensor_dim1[31:16]
                (64u << 16));                           // tile_dim0 = 64
  g1[4] = 32;                                           // tile_dim1 = 32
  g1[5] = rowlen;                                       // tensor_dim0_stride[31:0]
  g1[6] = 0;
  g1[7] = 0;
  const i32x4 z4 = {0, 0, 0, 0};
#if defined(__clang_major__) && (__clang_major__ >= 23)
  const i32x8 z8 = {0, 0, 0, 0, 0, 0, 0, 0};
  __builtin_amdgcn_tensor_load_to_lds(g0, g1, z4, z4, z8, 0);
#else
  __builtin_amdgcn_tensor_load_to_lds(g0, g1, z4, z4, 0);
#endif
}

// ---------------------------------------------------------------------------
// WMMA bf16 GEMM (B not transposed): C[M,N] = A[M,K] x B[K,N] + bias, opt ReLU.
// Block = 8 waves, one 64-wide column strip, 8 consecutive 16-row tiles
// (128 rows). The 32x64 f32 B tile per 32-wide K-step is block-shared:
// double-buffered in LDS, DMA'd by the Tensor Data Mover (wave 0 issues,
// s_wait_tensorcnt + barrier publishes, next tile's DMA overlaps the WMMAs).
// Each wave runs 4 x v_wmma_f32_16x16x32_bf16 per K-step sharing one A
// fragment. Requires M % 128 == 0 and N % 64 == 0 (true here).
// Fragment layouts per CDNA5 ISA:
//   A 16x32 bf16: lane m = lane&15; elems 0..7 -> k0+half*8..+7,
//                 elems 8..15 -> k0+16+half*8..+7
//   B 32x16 bf16: lane n = lane&15; elem e -> k = (e&7) + half*8 + (e>>3)*16 (same as A)
//   C/D f32:      vgpr r -> row = r + half*8, col = lane&15
// ---------------------------------------------------------------------------
template<bool RELU, bool HASBIAS>
__global__ __launch_bounds__(256) void wmma_gemm_tdm_kernel(
    const float* __restrict__ A, const float* __restrict__ Bm,
    const float* __restrict__ bias, float* __restrict__ C,
    int M, int K, int N)
{
  __shared__ float bs[2][32 * 64];

  const int waveId = threadIdx.x >> 5;
  const int lane   = threadIdx.x & 31;
  const int mgrp   = blockIdx.x % (M >> 7);      // 128-row group
  const int strip  = blockIdx.x / (M >> 7);      // 64-col strip
  const int tileM  = (mgrp << 3) + waveId;       // this wave's 16-row tile
  const int col0   = strip << 6;                 // first column of strip

  const int half = lane >> 4;
  const int l15  = lane & 15;
  const int rowA = (tileM << 4) + l15;
  const float* __restrict__ arow = A + (size_t)rowA * K;

  // Prime the pipeline: DMA B tile for k0 = 0 into buffer 0.
  if (waveId == 0)
    tdm_load_b_tile((unsigned)(uintptr_t)&bs[0][0], Bm + col0, N, K);

  v8f acc[4] = {{}, {}, {}, {}};
  const int steps = K >> 5;
  for (int s = 0; s < steps; ++s) {
    if (waveId == 0) __builtin_amdgcn_s_wait_tensorcnt(0);
    __syncthreads();                             // bs[s&1] visible to all
    if (waveId == 0 && (s + 1) < steps)          // overlap next DMA with WMMAs
      tdm_load_b_tile((unsigned)(uintptr_t)&bs[(s + 1) & 1][0],
                      Bm + (size_t)(s + 1) * 32 * N + col0, N, K);

    const int k0 = s << 5;
    // ---- A fragment: four coalesced float4 loads ----
    const float4* ap0 = (const float4*)(arow + k0 + (half << 3));
    const float4* ap1 = (const float4*)(arow + k0 + 16 + (half << 3));
    const float4 a0 = ap0[0], a1 = ap0[1], a2 = ap1[0], a3 = ap1[1];
    v16bf a;
    a[0]=(__bf16)a0.x; a[1]=(__bf16)a0.y; a[2]=(__bf16)a0.z; a[3]=(__bf16)a0.w;
    a[4]=(__bf16)a1.x; a[5]=(__bf16)a1.y; a[6]=(__bf16)a1.z; a[7]=(__bf16)a1.w;
    a[8]=(__bf16)a2.x; a[9]=(__bf16)a2.y; a[10]=(__bf16)a2.z; a[11]=(__bf16)a2.w;
    a[12]=(__bf16)a3.x; a[13]=(__bf16)a3.y; a[14]=(__bf16)a3.z; a[15]=(__bf16)a3.w;

    const float* __restrict__ bcur = &bs[s & 1][0];
#pragma unroll
    for (int j = 0; j < 4; ++j) {
      v16bf b;
#pragma unroll
      for (int e = 0; e < 16; ++e)
        b[e] = (__bf16)bcur[((e & 7) + (half << 3) + ((e >> 3) << 4)) * 64 + (j << 4) + l15];
      acc[j] = __builtin_amdgcn_wmma_f32_16x16x32_bf16(
          false, a, false, b, (short)0, acc[j], false, false);
    }
    __syncthreads();                             // all reads of bs[s&1] done
  }

#pragma unroll
  for (int j = 0; j < 4; ++j) {
    const int col = col0 + (j << 4) + l15;
    const float bsc = HASBIAS ? bias[col] : 0.0f;
#pragma unroll
    for (int r = 0; r < 8; ++r) {
      const int row = (tileM << 4) + r + (half << 3);
      float v = acc[j][r] + bsc;
      if (RELU) v = fmaxf(v, 0.0f);
      C[(size_t)row * N + col] = v;
    }
  }
}

// ---------------------------------------------------------------------------
// WMMA bf16 GEMM with transposed B: C[M,N] = A[M,K] x W^T (W is [N,K]).
// Per-lane B access is contiguous in K -> plain float4 global loads.
// One wave per 16x64 strip, 4 WMMAs per K-step sharing one A fragment.
// ---------------------------------------------------------------------------
__global__ __launch_bounds__(256) void wmma_gemm_trans_kernel(
    const float* __restrict__ A, const float* __restrict__ Bm,
    float* __restrict__ C, int M, int K, int N)
{
  const int wave   = blockIdx.x * (blockDim.x >> 5) + (threadIdx.x >> 5);
  const int lane   = threadIdx.x & 31;
  const int nstrip = N >> 6;
  const int tileM  = wave / nstrip;
  const int strip  = wave - tileM * nstrip;
  if (tileM >= (M >> 4)) return;                 // wave-uniform

  const int half  = lane >> 4;
  const int l15   = lane & 15;
  const int rowA  = (tileM << 4) + l15;
  const int colB0 = (strip << 6) + l15;
  const float* __restrict__ arow = A + (size_t)rowA * K;

  v8f acc[4] = {{}, {}, {}, {}};
  for (int k0 = 0; k0 < K; k0 += 32) {
    const float4* ap0 = (const float4*)(arow + k0 + (half << 3));
    const float4* ap1 = (const float4*)(arow + k0 + 16 + (half << 3));
    const float4 a0 = ap0[0], a1 = ap0[1], a2 = ap1[0], a3 = ap1[1];
    v16bf a;
    a[0]=(__bf16)a0.x; a[1]=(__bf16)a0.y; a[2]=(__bf16)a0.z; a[3]=(__bf16)a0.w;
    a[4]=(__bf16)a1.x; a[5]=(__bf16)a1.y; a[6]=(__bf16)a1.z; a[7]=(__bf16)a1.w;
    a[8]=(__bf16)a2.x; a[9]=(__bf16)a2.y; a[10]=(__bf16)a2.z; a[11]=(__bf16)a2.w;
    a[12]=(__bf16)a3.x; a[13]=(__bf16)a3.y; a[14]=(__bf16)a3.z; a[15]=(__bf16)a3.w;

#pragma unroll
    for (int j = 0; j < 4; ++j) {
      const int col = colB0 + (j << 4);
      const float4* bp = (const float4*)(Bm + (size_t)col * K + k0 + (half << 3));
      const float4 b0 = bp[0], b1 = bp[1], b2 = bp[4], b3 = bp[5];
      v16bf b;
      b[0]=(__bf16)b0.x; b[1]=(__bf16)b0.y; b[2]=(__bf16)b0.z; b[3]=(__bf16)b0.w;
      b[4]=(__bf16)b1.x; b[5]=(__bf16)b1.y; b[6]=(__bf16)b1.z; b[7]=(__bf16)b1.w;
      b[8]=(__bf16)b2.x; b[9]=(__bf16)b2.y; b[10]=(__bf16)b2.z; b[11]=(__bf16)b2.w;
      b[12]=(__bf16)b3.x; b[13]=(__bf16)b3.y; b[14]=(__bf16)b3.z; b[15]=(__bf16)b3.w;
      acc[j] = __builtin_amdgcn_wmma_f32_16x16x32_bf16(
          false, a, false, b, (short)0, acc[j], false, false);
    }
  }

#pragma unroll
  for (int j = 0; j < 4; ++j) {
    const int col = colB0 + (j << 4);
#pragma unroll
    for (int r = 0; r < 8; ++r) {
      const int row = (tileM << 4) + r + (half << 3);
      C[(size_t)row * N + col] = acc[j][r];
    }
  }
}

// ---------------------------------------------------------------------------
// c[b] = dot(bk, Qg[b]) — one wave per graph.
// ---------------------------------------------------------------------------
__global__ __launch_bounds__(256) void cdot_kernel(
    const float* __restrict__ Qg, const float* __restrict__ bk,
    float* __restrict__ cdot, int B)
{
  const int g    = blockIdx.x * (blockDim.x >> 5) + (threadIdx.x >> 5);
  const int lane = threadIdx.x & 31;
  if (g >= B) return;
  float acc = 0.0f;
#pragma unroll
  for (int i = 0; i < HDIM / 32; ++i)
    acc += Qg[(size_t)g * HDIM + lane + i * 32] * bk[lane + i * 32];
  acc = wave_reduce_sum(acc);
  if (lane == 0) cdot[g] = acc;
}

// ---------------------------------------------------------------------------
// ex[n] = exp((key_in[n] . T[batch[n]] + c[b]) / SCALE)
// One wave per node; float4 loads stream key_in (210 MB) fully coalesced.
// ---------------------------------------------------------------------------
__global__ __launch_bounds__(256) void score_kernel(
    const float* __restrict__ key_in, const float* __restrict__ T,
    const float* __restrict__ cdot, const int* __restrict__ batch,
    float* __restrict__ ex, int N)
{
  const int node = blockIdx.x * (blockDim.x >> 5) + (threadIdx.x >> 5);
  const int lane = threadIdx.x & 31;
  if (node >= N) return;
  const int b = batch[node];
  const float4* kr = (const float4*)(key_in + (size_t)node * HDIM) + lane;
  const float4* tr = (const float4*)(T + (size_t)b * HDIM) + lane;
  float acc = 0.0f;
#pragma unroll
  for (int i = 0; i < HDIM / 128; ++i) {   // 2 iters: 32 lanes * float4 = 128 floats
    const float4 kv = kr[i * 32];
    const float4 tv = tr[i * 32];
    acc += kv.x * tv.x + kv.y * tv.y + kv.z * tv.z + kv.w * tv.w;
  }
  acc = wave_reduce_sum(acc);
  if (lane == 0) ex[node] = expf((acc + cdot[b]) * INV_SCALE);
}

// ---------------------------------------------------------------------------
// One block (256 threads) per graph. batch is sorted, so a binary search
// gives this graph's node range [lo,hi). Compute softmax denom locally
// (no atomics), write attention, and accumulate agg[b] = sum attn*value_in.
// Streams value_in (210 MB) coalesced; global_prefetch hides HBM latency
// behind the short serial per-graph loop.
// ---------------------------------------------------------------------------
__global__ __launch_bounds__(256) void attn_agg_kernel(
    const float* __restrict__ ex, const float* __restrict__ value_in,
    const int* __restrict__ batch, float* __restrict__ attn_out,
    float* __restrict__ agg, int N)
{
  const int b = blockIdx.x;
  __shared__ int   srange[2];
  __shared__ float sred[8];
  __shared__ float sden;
  if (threadIdx.x == 0) {
    int lo = 0, hi = N;
    while (lo < hi) { int m = (lo + hi) >> 1; if (batch[m] <  b) lo = m + 1; else hi = m; }
    srange[0] = lo;
    hi = N;
    while (lo < hi) { int m = (lo + hi) >> 1; if (batch[m] <= b) lo = m + 1; else hi = m; }
    srange[1] = lo;
  }
  __syncthreads();
  const int lo = srange[0], hi = srange[1];

  float s = 0.0f;
  for (int n = lo + threadIdx.x; n < hi; n += 256) s += ex[n];
  s = wave_reduce_sum(s);
  if ((threadIdx.x & 31) == 0) sred[threadIdx.x >> 5] = s;
  __syncthreads();
  if (threadIdx.x == 0) {
    float t = 0.0f;
#pragma unroll
    for (int i = 0; i < 8; ++i) t += sred[i];
    sden = t;
  }
  __syncthreads();
  const float inv = (sden > 0.0f) ? (1.0f / sden) : 0.0f;

  const int h = threadIdx.x;               // one column per thread
  float a = 0.0f;
  for (int n = lo; n < hi; ++n) {          // ex[n] is a broadcast load
    if (n + 4 < hi)                        // -> global_prefetch_b8
      __builtin_prefetch(&value_in[(size_t)(n + 4) * HDIM + h], 0, 3);
    a += ex[n] * value_in[(size_t)n * HDIM + h];
  }
  agg[(size_t)b * HDIM + h] = a * inv;

  for (int n = lo + threadIdx.x; n < hi; n += 256) attn_out[n] = ex[n] * inv;
}

// ---------------------------------------------------------------------------
// out[row] = LayerNorm(x[row] + dy[row]) * g + b   — one block per row.
// ---------------------------------------------------------------------------
__global__ __launch_bounds__(256) void add_ln_kernel(
    const float* __restrict__ x, const float* __restrict__ dy,
    const float* __restrict__ g, const float* __restrict__ bb,
    float* __restrict__ out)
{
  const int row = blockIdx.x;
  const int h   = threadIdx.x;
  __shared__ float sred[8];
  __shared__ float sstat;
  const float v = x[(size_t)row * HDIM + h] + dy[(size_t)row * HDIM + h];

  float s = wave_reduce_sum(v);
  if ((h & 31) == 0) sred[h >> 5] = s;
  __syncthreads();
  if (h == 0) {
    float t = 0.0f;
#pragma unroll
    for (int i = 0; i < 8; ++i) t += sred[i];
    sstat = t * (1.0f / HDIM);
  }
  __syncthreads();
  const float mean = sstat;
  const float d = v - mean;

  float q = wave_reduce_sum(d * d);
  __syncthreads();                         // everyone has read sstat/sred
  if ((h & 31) == 0) sred[h >> 5] = q;
  __syncthreads();
  if (h == 0) {
    float t = 0.0f;
#pragma unroll
    for (int i = 0; i < 8; ++i) t += sred[i];
    sstat = t * (1.0f / HDIM);
  }
  __syncthreads();
  out[(size_t)row * HDIM + h] = d * rsqrtf(sstat + 1e-5f) * g[h] + bb[h];
}

// ---------------------------------------------------------------------------
extern "C" void kernel_launch(void* const* d_in, const int* in_sizes, int n_in,
                              void* d_out, int out_size, void* d_ws, size_t ws_size,
                              hipStream_t stream) {
  (void)n_in; (void)out_size; (void)ws_size;
  const float* smiles   = (const float*)d_in[0];
  const float* key_in   = (const float*)d_in[1];
  const float* value_in = (const float*)d_in[2];
  const int*   batch    = (const int*)  d_in[3];
  const float* wq       = (const float*)d_in[4];
  const float* bq       = (const float*)d_in[5];
  const float* wk       = (const float*)d_in[6];
  const float* bk       = (const float*)d_in[7];
  const float* wv       = (const float*)d_in[8];
  const float* bv       = (const float*)d_in[9];
  const float* ln0_g    = (const float*)d_in[10];
  const float* ln0_b    = (const float*)d_in[11];
  const float* ln1_g    = (const float*)d_in[12];
  const float* ln1_b    = (const float*)d_in[13];
  const float* w1       = (const float*)d_in[14];
  const float* c1       = (const float*)d_in[15];
  const float* w2       = (const float*)d_in[16];
  const float* c2       = (const float*)d_in[17];

  const int B = in_sizes[0] / HDIM;        // 4096
  const int N = in_sizes[3];               // 204800

  float* out_smiles = (float*)d_out;                 // [B,H]
  float* out_attn   = out_smiles + (size_t)B * HDIM; // [N]

  // Workspace layout (floats)
  float* ws   = (float*)d_ws;
  float* Qg   = ws;                         // B*H
  float* T    = Qg   + (size_t)B * HDIM;    // B*H
  float* cdot = T    + (size_t)B * HDIM;    // B
  float* ex   = cdot + B;                   // N
  float* agg  = ex   + N;                   // B*H
  float* dh   = agg  + (size_t)B * HDIM;    // B*H
  float* xln  = dh   + (size_t)B * HDIM;    // B*H
  float* h1   = xln  + (size_t)B * HDIM;    // B*4H
  float* ffn  = h1   + (size_t)B * HDIM4;   // B*H

  const int WAVES_PER_BLOCK = 8;            // 256 threads
  auto tdm_blocks = [](int M, int N_) {
    return (M >> 7) * (N_ >> 6);            // 128-row x 64-col block tiles
  };
  auto trans_blocks = [](int M, int N_) {
    const int strips = (M >> 4) * (N_ >> 6);
    return (strips + 7) / 8;
  };

  // 1) Qg = smiles @ wq + bq
  wmma_gemm_tdm_kernel<false, true><<<tdm_blocks(B, HDIM), 256, 0, stream>>>(
      smiles, wq, bq, Qg, B, HDIM, HDIM);
  // 2) T = Qg @ wk^T ; c = Qg . bk
  wmma_gemm_trans_kernel<<<trans_blocks(B, HDIM), 256, 0, stream>>>(
      Qg, wk, T, B, HDIM, HDIM);
  cdot_kernel<<<(B + WAVES_PER_BLOCK - 1) / WAVES_PER_BLOCK, 256, 0, stream>>>(
      Qg, bk, cdot, B);
  // 3) ex[n] = exp(score[n])   (streams key_in)
  score_kernel<<<(N + WAVES_PER_BLOCK - 1) / WAVES_PER_BLOCK, 256, 0, stream>>>(
      key_in, T, cdot, batch, ex, N);
  // 4) per-graph softmax + attention out + agg  (streams value_in)
  attn_agg_kernel<<<B, 256, 0, stream>>>(ex, value_in, batch, out_attn, agg, N);
  // 5) dh = agg @ wv + bv
  wmma_gemm_tdm_kernel<false, true><<<tdm_blocks(B, HDIM), 256, 0, stream>>>(
      agg, wv, bv, dh, B, HDIM, HDIM);
  // 6) xln = LN(smiles + dh)
  add_ln_kernel<<<B, 256, 0, stream>>>(smiles, dh, ln0_g, ln0_b, xln);
  // 7) h1 = relu(xln @ w1 + c1)
  wmma_gemm_tdm_kernel<true, true><<<tdm_blocks(B, HDIM4), 256, 0, stream>>>(
      xln, w1, c1, h1, B, HDIM, HDIM4);
  // 8) ffn = h1 @ w2 + c2
  wmma_gemm_tdm_kernel<false, true><<<tdm_blocks(B, HDIM), 256, 0, stream>>>(
      h1, w2, c2, ffn, B, HDIM4, HDIM);
  // 9) out = LN(xln + ffn)
  add_ln_kernel<<<B, 256, 0, stream>>>(xln, ffn, ln1_g, ln1_b, out_smiles);
}
